// I2GNN_2267742732766
// MI455X (gfx1250) — matrix operand order repaired and checked by reading, not compiled, not run on hardware
//
#include <hip/hip_runtime.h>
#include <cstdint>

#define NNODES 100000
#define NEDGES 1600000
#define CNSG2  20000
#define CNSG   2000
#define CNG    64
#define DD     64
#define LAYERS 5

typedef __attribute__((ext_vector_type(16))) _Float16 v16h;
typedef __attribute__((ext_vector_type(8)))  float    v8f;

__device__ __forceinline__ void atomAddF(float* p, float v) {
  __hip_atomic_fetch_add(p, v, __ATOMIC_RELAXED, __HIP_MEMORY_SCOPE_AGENT);
}

__global__ void k_zero(float* __restrict__ p, int n) {
  int i = blockIdx.x * blockDim.x + threadIdx.x;
  if (i < n) p[i] = 0.0f;
}

// Convert stacked f32 weights [nmat][K][N] -> transposed f16 [nmat][N][K]
__global__ void k_cvt_t(const float* __restrict__ W, _Float16* __restrict__ Wt,
                        int K, int Nw, int total) {
  int i = blockIdx.x * blockDim.x + threadIdx.x;
  if (i >= total) return;
  int kn = K * Nw;
  int mat = i / kn, r = i - mat * kn;
  int k = r / Nw, n = r - k * Nw;
  Wt[(size_t)mat * kn + (size_t)n * K + k] = (_Float16)W[i];
}

// ze[n,d] = z_emb[z[n,0],d] + z_emb[z[n,1],d]
__global__ void k_embed(const int* __restrict__ z, const float* __restrict__ emb,
                        float* __restrict__ ze, int n) {
  int i = blockIdx.x * blockDim.x + threadIdx.x;
  if (i >= n) return;
  int node = i >> 6, d = i & 63;
  int z0 = z[2 * node], z1 = z[2 * node + 1];
  ze[i] = emb[z0 * DD + d] + emb[z1 * DD + d];
}

// m[dst[e], q..q+3] += h[src[e], q..q+3]   (n = E*16 work items)
__global__ void k_scatter4(const float* __restrict__ h, const int* __restrict__ src,
                           const int* __restrict__ dst, float* __restrict__ m, int n) {
  int i = blockIdx.x * blockDim.x + threadIdx.x;
  if (i >= n) return;
  int e = i >> 4, q = (i & 15) << 2;
  const float4 v = *reinterpret_cast<const float4*>(&h[(size_t)src[e] * DD + q]);
  float* o = &m[(size_t)dst[e] * DD + q];
  atomAddF(o + 0, v.x); atomAddF(o + 1, v.y);
  atomAddF(o + 2, v.z); atomAddF(o + 3, v.w);
}

// out[map[r], q..q+3] += in[r, q..q+3]   (n = cnt*16 work items)
__global__ void k_pool4(const float* __restrict__ in, const int* __restrict__ map,
                        float* __restrict__ out, int n) {
  int i = blockIdx.x * blockDim.x + threadIdx.x;
  if (i >= n) return;
  int r = i >> 4, q = (i & 15) << 2;
  const float4 v = *reinterpret_cast<const float4*>(&in[(size_t)r * DD + q]);
  float* o = &out[(size_t)map[r] * DD + q];
  atomAddF(o + 0, v.x); atomAddF(o + 1, v.y);
  atomAddF(o + 2, v.z); atomAddF(o + 3, v.w);
}

// GRU combine: gi/gh are [N,192] with (r,z,n) blocks of 64
__global__ void k_gru(const float* __restrict__ gi, const float* __restrict__ gh,
                      const float* __restrict__ x, float* __restrict__ xo, int n) {
  int i = blockIdx.x * blockDim.x + threadIdx.x;
  if (i >= n) return;
  int node = i >> 6, d = i & 63;
  size_t b = (size_t)node * 192 + d;
  float ir = gi[b], iz = gi[b + 64], in2 = gi[b + 128];
  float hr = gh[b], hz = gh[b + 64], hn = gh[b + 128];
  float r  = 1.0f / (1.0f + expf(-(ir + hr)));
  float zg = 1.0f / (1.0f + expf(-(iz + hz)));
  float nn = tanhf(in2 + r * hn);
  xo[i] = (1.0f - zg) * nn + zg * x[i];
}

// C[M, NT*16] = act(cat(A,A2)[M,K] @ B[K,NT*16] + bias)
// Bt is the pre-transposed f16 weight [NT*16][K]; staged to LDS per block.
// One wave computes a full 16-row stripe (NT accumulator tiles), reusing each
// A fragment across NT WMMAs. K-loop fully unrolled (K compile-time).
template<int K, int NT>
__global__ __launch_bounds__(256)
void k_gemm_lds(const float* __restrict__ A,
                const float* __restrict__ A2,           // concat part (cols >= 64), may be null
                const _Float16* __restrict__ Bt,        // [NT*16][K] f16
                const float* __restrict__ bias,
                float* __restrict__ C,
                int M, int act) {
  constexpr int NOUT = NT * 16;
  __shared__ __align__(32) _Float16 Bs[NOUT * K];

  // Stage full transposed weight into LDS (coalesced b128 -> ds_store_b128)
  {
    constexpr int NVEC = NOUT * K * (int)sizeof(_Float16) / 16;
    const uint4* gsrc = reinterpret_cast<const uint4*>(Bt);
    uint4* ldst = reinterpret_cast<uint4*>(Bs);
    for (int j = threadIdx.x; j < NVEC; j += 256) ldst[j] = gsrc[j];
  }
  __syncthreads();

  int tile = (blockIdx.x * blockDim.x + threadIdx.x) >> 5;
  if (tile >= (M >> 4)) return;
  int m0 = tile << 4;
  int lane = threadIdx.x & 31;
  int hi   = lane >> 4;                 // half-wave selector
  int rowA = m0 + (lane & 15);
  int kbA  = hi << 3;                   // A: 0 or 8
  int lcol = lane & 15;
  int kbB  = hi << 3;                   // B: 0 or 8

  v8f zero = {0.f, 0.f, 0.f, 0.f, 0.f, 0.f, 0.f, 0.f};
  v8f acc[NT];
#pragma unroll
  for (int t = 0; t < NT; ++t) acc[t] = zero;

#pragma unroll
  for (int kk = 0; kk < K; kk += 32) {
    // A fragment: two contiguous 8-float segments (vector loads + cvt to f16)
    v16h a;
    int s0 = kk + kbA;
    const float* p0 = (A2 != nullptr && s0 >= DD) ? (A2 + (size_t)rowA * DD + (s0 - DD))
                                                  : (A + (size_t)rowA * DD + s0);
    float4 f0 = *reinterpret_cast<const float4*>(p0);
    float4 f1 = *reinterpret_cast<const float4*>(p0 + 4);
    a[0]  = (_Float16)f0.x; a[1]  = (_Float16)f0.y; a[2]  = (_Float16)f0.z; a[3]  = (_Float16)f0.w;
    a[4]  = (_Float16)f1.x; a[5]  = (_Float16)f1.y; a[6]  = (_Float16)f1.z; a[7]  = (_Float16)f1.w;
    int s1 = kk + 16 + kbA;
    const float* p1 = (A2 != nullptr && s1 >= DD) ? (A2 + (size_t)rowA * DD + (s1 - DD))
                                                  : (A + (size_t)rowA * DD + s1);
    float4 g0 = *reinterpret_cast<const float4*>(p1);
    float4 g1 = *reinterpret_cast<const float4*>(p1 + 4);
    a[8]  = (_Float16)g0.x; a[9]  = (_Float16)g0.y; a[10] = (_Float16)g0.z; a[11] = (_Float16)g0.w;
    a[12] = (_Float16)g1.x; a[13] = (_Float16)g1.y; a[14] = (_Float16)g1.z; a[15] = (_Float16)g1.w;

    // NT B-fragments from LDS (two 16-byte segments each), NT WMMAs reuse `a`
#pragma unroll
    for (int t = 0; t < NT; ++t) {
      typedef __attribute__((ext_vector_type(8))) _Float16 v8h_;
      const v8h_ b0 = *reinterpret_cast<const v8h_*>(&Bs[(size_t)(t * 16 + lcol) * K + kk + kbB]);
      const v8h_ b1 = *reinterpret_cast<const v8h_*>(&Bs[(size_t)(t * 16 + lcol) * K + kk + 16 + kbB]);
      const v16h b = __builtin_shufflevector(b0, b1, 0, 1, 2, 3, 4, 5, 6, 7, 8, 9, 10, 11, 12, 13, 14, 15);
      acc[t] = __builtin_amdgcn_wmma_f32_16x16x32_f16(false, a, false, b, (short)0,
                                                      acc[t], false, false);
    }
  }

  int rbase = m0 + (hi << 3);
#pragma unroll
  for (int t = 0; t < NT; ++t) {
    int col = t * 16 + lcol;
    float bv = (bias != nullptr) ? bias[col] : 0.0f;
#pragma unroll
    for (int r = 0; r < 8; ++r) {
      float v = acc[t][r] + bv;
      if (act == 1) v = fmaxf(v, 0.0f);
      C[(size_t)(rbase + r) * NOUT + col] = v;
    }
  }
}

// Tiny head: [64,64] -> elu fc1(32) -> elu fc2(16) -> fc3(1)
__global__ void k_head(const float* __restrict__ x,
                       const float* __restrict__ W1, const float* __restrict__ b1,
                       const float* __restrict__ W2, const float* __restrict__ b2,
                       const float* __restrict__ W3, const float* __restrict__ b3,
                       float* __restrict__ out) {
  int g = threadIdx.x;
  if (g >= CNG) return;
  float h1[32];
  for (int j = 0; j < 32; ++j) {
    float s = b1[j];
    for (int k = 0; k < 64; ++k) s += x[g * 64 + k] * W1[k * 32 + j];
    h1[j] = (s > 0.0f) ? s : expm1f(s);
  }
  float h2[16];
  for (int j = 0; j < 16; ++j) {
    float s = b2[j];
    for (int k = 0; k < 32; ++k) s += h1[k] * W2[k * 16 + j];
    h2[j] = (s > 0.0f) ? s : expm1f(s);
  }
  float s = b3[0];
  for (int k = 0; k < 16; ++k) s += h2[k] * W3[k];
  out[g] = s;
}

static inline int cdiv_i(long a, int b) { return (int)((a + b - 1) / b); }

extern "C" void kernel_launch(void* const* d_in, const int* in_sizes, int n_in,
                              void* d_out, int out_size, void* d_ws, size_t ws_size,
                              hipStream_t stream) {
  (void)in_sizes; (void)n_in; (void)out_size; (void)ws_size;

  const int*   z     = (const int*)d_in[0];
  const int*   ei    = (const int*)d_in[1];
  const int*   src   = ei;
  const int*   dst   = ei + NEDGES;
  const int*   n2s2  = (const int*)d_in[2];
  const int*   s2s   = (const int*)d_in[3];
  const int*   s2g   = (const int*)d_in[4];
  const float* z_emb = (const float*)d_in[5];
  const float* trW   = (const float*)d_in[6];
  const float* trB   = (const float*)d_in[7];
  const float* cvW   = (const float*)d_in[8];
  const float* Wih   = (const float*)d_in[9];
  const float* bih   = (const float*)d_in[10];
  const float* Whh   = (const float*)d_in[11];
  const float* bhh   = (const float*)d_in[12];
  const float* epW1  = (const float*)d_in[13];
  const float* epb1  = (const float*)d_in[14];
  const float* epW2  = (const float*)d_in[15];
  const float* epb2  = (const float*)d_in[16];
  const float* npW1  = (const float*)d_in[17];
  const float* npb1  = (const float*)d_in[18];
  const float* npW2  = (const float*)d_in[19];
  const float* npb2  = (const float*)d_in[20];
  const float* fc1W  = (const float*)d_in[21];
  const float* fc1b  = (const float*)d_in[22];
  const float* fc2W  = (const float*)d_in[23];
  const float* fc2b  = (const float*)d_in[24];
  const float* fc3W  = (const float*)d_in[25];
  const float* fc3b  = (const float*)d_in[26];

  // ---- Workspace layout (f32 region then f16 weight region) ----
  float* ws  = (float*)d_ws;
  size_t N64 = (size_t)NNODES * DD;
  float* xA  = ws;
  float* xB  = xA + N64;
  float* ze  = xB + N64;
  float* h   = ze + N64;
  float* m   = h + N64;
  float* gi  = m + N64;
  float* gh  = gi + (size_t)NNODES * 192;
  float* p1  = gh + (size_t)NNODES * 192;
  float* p1t = p1 + (size_t)CNSG2 * DD;
  float* p2  = p1t + (size_t)CNSG2 * DD;
  float* p2t = p2 + (size_t)CNSG * DD;
  float* p3  = p2t + (size_t)CNSG * DD;
  _Float16* f16base = (_Float16*)(p3 + (size_t)CNG * DD);
  _Float16* trWt  = f16base;                               // 4 x [64][128]
  _Float16* cvWt  = trWt + (size_t)4 * 128 * 64;           // 5 x [64][64]
  _Float16* WihT  = cvWt + (size_t)5 * 64 * 64;            // 5 x [192][64]
  _Float16* WhhT  = WihT + (size_t)5 * 64 * 192;           // 5 x [192][64]
  _Float16* ep1T  = WhhT + (size_t)5 * 64 * 192;
  _Float16* ep2T  = ep1T + (size_t)64 * 64;
  _Float16* np1T  = ep2T + (size_t)64 * 64;
  _Float16* np2T  = np1T + (size_t)64 * 64;

  const dim3 blk(256);
  const int nND = NNODES * DD;       // 6.4M
  const int nE4 = NEDGES * 16;       // 25.6M float4 quads

  // ---- Weight convert + transpose (tiny, once per launch) ----
  auto cvt = [&](const float* W, _Float16* Wt, int K, int Nw, int nmat) {
    int total = nmat * K * Nw;
    hipLaunchKernelGGL(k_cvt_t, dim3(cdiv_i(total, 256)), blk, 0, stream, W, Wt, K, Nw, total);
  };
  cvt(trW, trWt, 128, 64, 4);
  cvt(cvW, cvWt, 64, 64, 5);
  cvt(Wih, WihT, 64, 192, 5);
  cvt(Whh, WhhT, 64, 192, 5);
  cvt(epW1, ep1T, 64, 64, 1);
  cvt(epW2, ep2T, 64, 64, 1);
  cvt(npW1, np1T, 64, 64, 1);
  cvt(npW2, np2T, 64, 64, 1);

  auto blocks_for = [](int M) { return dim3((unsigned)cdiv_i((M >> 4), 8)); };

  const float* x = nullptr;
  float* bufs[2] = {xA, xB};
  int cur = 0;

  for (int layer = 0; layer < LAYERS; ++layer) {
    hipLaunchKernelGGL(k_embed, dim3(cdiv_i(nND, 256)), blk, 0, stream,
                       z, z_emb + (size_t)layer * 100 * DD, ze, nND);
    if (layer == 0) {
      x = ze;
    } else {
      float* xt = bufs[cur];
      hipLaunchKernelGGL((k_gemm_lds<128, 4>), blocks_for(NNODES), blk, 0, stream,
                         x, ze, trWt + (size_t)(layer - 1) * 128 * 64,
                         trB + (size_t)(layer - 1) * DD, xt, NNODES, 0);
      x = xt; cur ^= 1;
    }
    // h = x @ conv_W[layer]
    hipLaunchKernelGGL((k_gemm_lds<64, 4>), blocks_for(NNODES), blk, 0, stream,
                       x, (const float*)nullptr, cvWt + (size_t)layer * 64 * 64,
                       (const float*)nullptr, h, NNODES, 0);
    // m = segment_sum(h[src], dst)
    hipLaunchKernelGGL(k_zero, dim3(cdiv_i(nND, 256)), blk, 0, stream, m, nND);
    hipLaunchKernelGGL(k_scatter4, dim3(cdiv_i(nE4, 256)), blk, 0, stream,
                       h, src, dst, m, nE4);
    // GRU gates
    hipLaunchKernelGGL((k_gemm_lds<64, 12>), blocks_for(NNODES), blk, 0, stream,
                       m, (const float*)nullptr, WihT + (size_t)layer * 64 * 192,
                       bih + (size_t)layer * 192, gi, NNODES, 0);
    hipLaunchKernelGGL((k_gemm_lds<64, 12>), blocks_for(NNODES), blk, 0, stream,
                       x, (const float*)nullptr, WhhT + (size_t)layer * 64 * 192,
                       bhh + (size_t)layer * 192, gh, NNODES, 0);
    float* xn = bufs[cur];
    hipLaunchKernelGGL(k_gru, dim3(cdiv_i(nND, 256)), blk, 0, stream,
                       gi, gh, x, xn, nND);
    x = xn; cur ^= 1;
  }

  // Pool N -> NSG2, MLP
  int n1 = CNSG2 * DD;
  hipLaunchKernelGGL(k_zero, dim3(cdiv_i(n1, 256)), blk, 0, stream, p1, n1);
  hipLaunchKernelGGL(k_pool4, dim3(cdiv_i(NNODES * 16, 256)), blk, 0, stream,
                     x, n2s2, p1, NNODES * 16);
  hipLaunchKernelGGL((k_gemm_lds<64, 4>), blocks_for(CNSG2), blk, 0, stream,
                     p1, (const float*)nullptr, ep1T, epb1, p1t, CNSG2, 1);
  hipLaunchKernelGGL((k_gemm_lds<64, 4>), blocks_for(CNSG2), blk, 0, stream,
                     p1t, (const float*)nullptr, ep2T, epb2, p1, CNSG2, 0);

  // Pool NSG2 -> NSG, MLP
  int n2 = CNSG * DD;
  hipLaunchKernelGGL(k_zero, dim3(cdiv_i(n2, 256)), blk, 0, stream, p2, n2);
  hipLaunchKernelGGL(k_pool4, dim3(cdiv_i(CNSG2 * 16, 256)), blk, 0, stream,
                     p1, s2s, p2, CNSG2 * 16);
  hipLaunchKernelGGL((k_gemm_lds<64, 4>), blocks_for(CNSG), blk, 0, stream,
                     p2, (const float*)nullptr, np1T, npb1, p2t, CNSG, 1);
  hipLaunchKernelGGL((k_gemm_lds<64, 4>), blocks_for(CNSG), blk, 0, stream,
                     p2t, (const float*)nullptr, np2T, npb2, p2, CNSG, 0);

  // Pool NSG -> NG
  int n3 = CNG * DD;
  hipLaunchKernelGGL(k_zero, dim3(cdiv_i(n3, 256)), blk, 0, stream, p3, n3);
  hipLaunchKernelGGL(k_pool4, dim3(cdiv_i(CNSG * 16, 256)), blk, 0, stream,
                     p2, s2g, p3, CNSG * 16);

  // Head
  hipLaunchKernelGGL(k_head, dim3(1), dim3(64), 0, stream,
                     p3, fc1W, fc1b, fc2W, fc2b, fc3W, fc3b, (float*)d_out);
}
